// OptimizedHybridQNN_21449066676263
// MI455X (gfx1250) — matrix operand order repaired and checked by reading, not compiled, not run on hardware
//
#include <hip/hip_runtime.h>
#include <hip/hip_bf16.h>
#include <math.h>

// ---------------------------------------------------------------- types
typedef __attribute__((ext_vector_type(16))) _Float16 v16h;
typedef __attribute__((ext_vector_type(8)))  _Float16 v8h;
typedef __attribute__((ext_vector_type(8)))  float    v8f;

union vh16 { v16h v; v8h h[2]; };

// ---------------------------------------------------------------- ws layout (floats)
// stats region (256 floats):
#define S1S   0     // conv1 sum[4]
#define S1Q   4     // conv1 sumsq[4]
#define S2S   8     // conv2 sum[8]
#define S2Q   16
#define S3S   24    // conv3 sum[16]
#define S3Q   40
#define SPS   56    // bnp sum[32]
#define SPQ   88
#define SFS   120   // bnf sum[32]
#define SFQ   152

#define OFF_C1  256                      // 4096*4*14*14 = 3211264
#define OFF_C2  (OFF_C1 + 3211264)       // 4096*8*7*7  = 1605632
#define OFF_C3  (OFF_C2 + 1605632)       // 4096*16*3*3 = 589824
#define OFF_CF  (OFF_C3 + 589824)        // cnn_feat 4096*32
#define OFF_QI  (OFF_CF + 131072)        // q_in 4096*256
#define OFF_QR  (OFF_QI + 1048576)       // q_raw 4096*4
#define OFF_H2  (OFF_QR + 16384)         // h2pre 4096*32
#define OFF_F1  (OFF_H2 + 131072)        // f1 4096*32

#define EPS_BN 1e-5f

// ---------------------------------------------------------------- WMMA helpers
// A fragment: 16x32 f16, A row-major in LDS with stride lda (lda, k0 multiples of 8).
// lane L: M = L%16 ; halves e=0..15 -> K = k0 + hi*16 + e
__device__ __forceinline__ v16h hq_load_a(const _Float16* A, int lda, int k0, int lane) {
  int m = lane & 15, hi = (lane >> 4) & 1;
  const _Float16* p = A + m * lda + k0 + hi * 16;
  vh16 u;
  u.h[0] = *(const v8h*)p;
  u.h[1] = *(const v8h*)(p + 8);
  return u.v;
}
// B fragment from fragment-native (swizzled) LDS buffer:
// slot layout: Bf[frag*512 + L*16 + e] == B[kbase(L)+e][n(L)]
__device__ __forceinline__ v16h hq_load_bf(const _Float16* Bf, int frag, int lane) {
  return *(const v16h*)(Bf + frag * 512 + lane * 16);
}
__device__ __forceinline__ v8f hq_wmma(v16h a, v16h b, v8f c) {
  return __builtin_amdgcn_wmma_f32_16x16x32_f16(false, a, false, b, (short)0, c, false, false);
}
// Swizzled fill of one fragment-lane slot (16 contiguous K values of one column n).
// src is row-major [N][K] (weights stored as (out,in)); kbase..kbase+15, bounds-padded.
__device__ __forceinline__ void hq_fill_slot(_Float16* Bf, int slot, const float* src,
                                             int ldsrc, int n, int kbase, int K) {
  vh16 u;
#pragma unroll
  for (int e = 0; e < 16; ++e) {
    int k = kbase + e;
    u.v[e] = (k < K) ? (_Float16)src[n * ldsrc + k] : (_Float16)0.f;
  }
  *(v16h*)(Bf + slot * 16) = u.v;
}

// ---------------------------------------------------------------- K0: zero stats
__global__ void hq_zero_stats(float* stats) {
  if (threadIdx.x < 256) stats[threadIdx.x] = 0.f;
}

// ---------------------------------------------------------------- K1: conv1 (1->4, s2 p1) + bn1 stats
__global__ __launch_bounds__(256) void hq_conv1(const float* __restrict__ x,
                                                const float* __restrict__ w,
                                                const float* __restrict__ b,
                                                float* __restrict__ out, float* stats) {
  __shared__ float sS[4], sQ[4];
  int tid = threadIdx.x;
  if (tid < 4) { sS[tid] = 0.f; sQ[tid] = 0.f; }
  __syncthreads();
  int idx = blockIdx.x * 256 + tid;              // 3,211,264 total (exact grid)
  int ow = idx % 14, t = idx / 14;
  int oh = t % 14;  t /= 14;
  int co = t & 3;   int n = t >> 2;
  const float* xin = x + n * 784;
  __builtin_prefetch(xin, 0, 1);
  float acc = b[co];
#pragma unroll
  for (int kh = 0; kh < 3; ++kh)
#pragma unroll
    for (int kw = 0; kw < 3; ++kw) {
      int ih = 2 * oh + kh - 1, iw = 2 * ow + kw - 1;
      if ((unsigned)ih < 28u && (unsigned)iw < 28u)
        acc = fmaf(w[co * 9 + kh * 3 + kw], xin[ih * 28 + iw], acc);
    }
  out[idx] = acc;
  atomicAdd(&sS[co], acc);
  atomicAdd(&sQ[co], acc * acc);
  __syncthreads();
  if (tid < 4) { atomicAdd(&stats[S1S + tid], sS[tid]); atomicAdd(&stats[S1Q + tid], sQ[tid]); }
}

// ---------------------------------------------------------------- K2: bn1+relu fused into conv2 (4->8, s2 p1) + bn2 stats
__global__ __launch_bounds__(256) void hq_conv2(const float* __restrict__ c1,
                                                const float* __restrict__ w,
                                                const float* __restrict__ b,
                                                const float* __restrict__ g1,
                                                const float* __restrict__ bb1,
                                                float* __restrict__ out, float* stats) {
  __shared__ float sc[4], sh[4], sS[8], sQ[8];
  int tid = threadIdx.x;
  if (tid < 4) {
    float N = 4096.f * 196.f;
    float m = stats[S1S + tid] / N;
    float v = stats[S1Q + tid] / N - m * m;
    float s = g1[tid] * rsqrtf(v + EPS_BN);
    sc[tid] = s; sh[tid] = bb1[tid] - m * s;
  }
  if (tid < 8) { sS[tid] = 0.f; sQ[tid] = 0.f; }
  __syncthreads();
  int idx = blockIdx.x * 256 + tid;              // 1,605,632 total
  int ow = idx % 7, t = idx / 7;
  int oh = t % 7;  t /= 7;
  int co = t & 7;  int n = t >> 3;
  float acc = b[co];
#pragma unroll
  for (int ci = 0; ci < 4; ++ci)
#pragma unroll
    for (int kh = 0; kh < 3; ++kh)
#pragma unroll
      for (int kw = 0; kw < 3; ++kw) {
        int ih = 2 * oh + kh - 1, iw = 2 * ow + kw - 1;
        if ((unsigned)ih < 14u && (unsigned)iw < 14u) {
          float val = fmaxf(fmaf(c1[((n * 4 + ci) * 14 + ih) * 14 + iw], sc[ci], sh[ci]), 0.f);
          acc = fmaf(w[((co * 4 + ci) * 3 + kh) * 3 + kw], val, acc);
        }
      }
  out[idx] = acc;
  atomicAdd(&sS[co], acc);
  atomicAdd(&sQ[co], acc * acc);
  __syncthreads();
  if (tid < 8) { atomicAdd(&stats[S2S + tid], sS[tid]); atomicAdd(&stats[S2Q + tid], sQ[tid]); }
}

// ---------------------------------------------------------------- K3: bn2+relu fused into conv3 (8->16, s2 p0) + bn3 stats
__global__ __launch_bounds__(256) void hq_conv3(const float* __restrict__ c2,
                                                const float* __restrict__ w,
                                                const float* __restrict__ b,
                                                const float* __restrict__ g2,
                                                const float* __restrict__ bb2,
                                                float* __restrict__ out, float* stats) {
  __shared__ float sc[8], sh[8], sS[16], sQ[16];
  int tid = threadIdx.x;
  if (tid < 8) {
    float N = 4096.f * 49.f;
    float m = stats[S2S + tid] / N;
    float v = stats[S2Q + tid] / N - m * m;
    float s = g2[tid] * rsqrtf(v + EPS_BN);
    sc[tid] = s; sh[tid] = bb2[tid] - m * s;
  }
  if (tid < 16) { sS[tid] = 0.f; sQ[tid] = 0.f; }
  __syncthreads();
  int idx = blockIdx.x * 256 + tid;              // 589,824 total
  int ow = idx % 3, t = idx / 3;
  int oh = t % 3;  t /= 3;
  int co = t & 15; int n = t >> 4;
  float acc = b[co];
#pragma unroll
  for (int ci = 0; ci < 8; ++ci)
#pragma unroll
    for (int kh = 0; kh < 3; ++kh)
#pragma unroll
      for (int kw = 0; kw < 3; ++kw) {
        int ih = 2 * oh + kh, iw = 2 * ow + kw;   // pad 0, always in [0,6]
        float val = fmaxf(fmaf(c2[((n * 8 + ci) * 7 + ih) * 7 + iw], sc[ci], sh[ci]), 0.f);
        acc = fmaf(w[((co * 8 + ci) * 3 + kh) * 3 + kw], val, acc);
      }
  out[idx] = acc;
  atomicAdd(&sS[co], acc);
  atomicAdd(&sQ[co], acc * acc);
  __syncthreads();
  if (tid < 16) { atomicAdd(&stats[S3S + tid], sS[tid]); atomicAdd(&stats[S3Q + tid], sQ[tid]); }
}

// ---------------------------------------------------------------- K4: bn3+relu -> WMMA GEMM (4096x144)@(144x32) -> relu cnn_feat
// B prepacked in fragment order: frag = kc*2 + nt (kc=0..4, nt=0..1)
__global__ __launch_bounds__(128) void hq_cnn_head(const float* __restrict__ c3,
                                                   const float* __restrict__ stats,
                                                   const float* __restrict__ g3,
                                                   const float* __restrict__ b3,
                                                   const float* __restrict__ fcw,
                                                   const float* __restrict__ fcb,
                                                   float* __restrict__ cnn_feat) {
  __shared__ __align__(32) _Float16 sB[10 * 512];     // 10 fragments, 10KB
  __shared__ __align__(32) _Float16 sA[4][16 * 160];  // per-wave 16 rows x 160 K
  __shared__ float sScale[16], sShift[16];
  int tid = threadIdx.x, wave = tid >> 5, lane = tid & 31;
  if (tid < 16) {
    float N = 4096.f * 9.f;
    float m = stats[S3S + tid] / N;
    float v = stats[S3Q + tid] / N - m * m;
    float s = g3[tid] * rsqrtf(v + EPS_BN);
    sScale[tid] = s; sShift[tid] = b3[tid] - m * s;
  }
  // swizzled fill: slot i = frag*32 + L
  for (int i = tid; i < 10 * 32; i += 128) {
    int frag = i >> 5, L = i & 31;
    int kc = frag >> 1, nt = frag & 1;
    int kbase = kc * 32 + ((L >> 4) & 1) * 16;
    int n = nt * 16 + (L & 15);
    hq_fill_slot(sB, i, fcw, 144, n, kbase, 144);
  }
  __syncthreads();
  int row0 = blockIdx.x * 64 + wave * 16;
  for (int i = lane; i < 16 * 160; i += 32) {
    int m = i / 160, k = i % 160;
    float val = 0.f;
    if (k < 144) {
      int c = k / 9;
      val = fmaxf(fmaf(c3[(row0 + m) * 144 + k], sScale[c], sShift[c]), 0.f);
    }
    sA[wave][m * 160 + k] = (_Float16)val;
  }
  __syncthreads();
  v8f c0 = {}, c1 = {};
#pragma unroll
  for (int kc = 0; kc < 5; ++kc) {
    v16h a  = hq_load_a(&sA[wave][0], 160, kc * 32, lane);
    v16h b0 = hq_load_bf(sB, kc * 2 + 0, lane);
    c0 = hq_wmma(a, b0, c0);
    v16h b1 = hq_load_bf(sB, kc * 2 + 1, lane);
    c1 = hq_wmma(a, b1, c1);
  }
  int n = lane & 15, hi = lane >> 4;
#pragma unroll
  for (int r = 0; r < 8; ++r) {
    int m = r + 8 * hi;
    cnn_feat[(row0 + m) * 32 + n]      = fmaxf(c0[r] + fcb[n], 0.f);
    cnn_feat[(row0 + m) * 32 + 16 + n] = fmaxf(c1[r] + fcb[16 + n], 0.f);
  }
}

// ---------------------------------------------------------------- K5: pre1 GEMM (WMMA) -> relu -> pre2 GEMM (WMMA) -> tanh -> L2 normalize
__global__ __launch_bounds__(256) void hq_pre_path(const float* __restrict__ x,
                                                   const float* __restrict__ w1,
                                                   const float* __restrict__ b1,
                                                   const float* __restrict__ w2,
                                                   const float* __restrict__ b2,
                                                   float* __restrict__ q_in) {
  __shared__ __align__(32) _Float16 sA[16 * 800];   // 25.6KB; reused as sQ (floats) in phase 3
  __shared__ __align__(32) _Float16 sB[25 * 512];   // 25.6KB; phase1: 25 frags; phase2: 16 frags
  __shared__ float sPart[8 * 256];                  // per-wave partial C tiles
  __shared__ float sH1[256];                        // reduced h1 (16x16)
  __shared__ __align__(32) _Float16 sA2[16 * 32];
  __shared__ float sNrm[16];
  int tid = threadIdx.x, wave = tid >> 5, lane = tid & 31;
  int row0 = blockIdx.x * 16;
  // A = x rows as f16, K 784 padded to 800
  for (int i = tid; i < 16 * 800; i += 256) {
    int m = i / 800, k = i % 800;
    sA[i] = (k < 784) ? (_Float16)x[(row0 + m) * 784 + k] : (_Float16)0.f;
  }
  // B1 = pre1_w^T in fragment order: frag = kc (n-tile is single)
  for (int i = tid; i < 25 * 32; i += 256) {
    int kc = i >> 5, L = i & 31;
    int kbase = kc * 32 + ((L >> 4) & 1) * 16;
    hq_fill_slot(sB, i, w1, 784, L & 15, kbase, 784);
  }
  __syncthreads();
  // split K chunks (25) across 8 waves, fp32 accumulate
  v8f c = {};
  for (int kc = wave; kc < 25; kc += 8) {
    v16h a = hq_load_a(sA, 800, kc * 32, lane);
    v16h b = hq_load_bf(sB, kc, lane);
    c = hq_wmma(a, b, c);
  }
  {
    int n = lane & 15, hi = lane >> 4;
#pragma unroll
    for (int r = 0; r < 8; ++r) sPart[wave * 256 + (r + 8 * hi) * 16 + n] = c[r];
  }
  __syncthreads();
  // deterministic cross-wave reduction of partials
  {
    float s = 0.f;
#pragma unroll
    for (int w = 0; w < 8; ++w) s += sPart[w * 256 + tid];
    sH1[tid] = s;
  }
  __syncthreads();
  // A2 = relu(h1 + b1) as f16, K 16 padded to 32
  for (int i = tid; i < 16 * 32; i += 256) {
    int m = i >> 5, k = i & 31;
    sA2[i] = (k < 16) ? (_Float16)fmaxf(sH1[m * 16 + k] + b1[k], 0.f) : (_Float16)0.f;
  }
  __syncthreads();
  // B2 = pre2_w^T in fragment order: frag = n-tile 0..15 (single K chunk, K=16 padded)
  for (int i = tid; i < 16 * 32; i += 256) {
    int nt = i >> 5, L = i & 31;
    int kbase = ((L >> 4) & 1) * 16;
    hq_fill_slot(sB, i, w2, 16, nt * 16 + (L & 15), kbase, 16);
  }
  __syncthreads();
  float* sQ = (float*)sA;                           // overlay: 16x256 floats (16KB < 25.6KB)
  v16h a2 = hq_load_a(sA2, 32, 0, lane);
#pragma unroll
  for (int t = 0; t < 2; ++t) {
    int nt = wave * 2 + t;
    int n0 = nt * 16;
    v16h b = hq_load_bf(sB, nt, lane);
    v8f cc = {};
    cc = hq_wmma(a2, b, cc);
    int n = lane & 15, hi = lane >> 4;
#pragma unroll
    for (int r = 0; r < 8; ++r) {
      int m = r + 8 * hi;
      sQ[m * 256 + n0 + n] = tanhf(cc[r] + b2[n0 + n]);
    }
  }
  __syncthreads();
  if (tid < 16) {
    float s = 0.f;
    for (int j = 0; j < 256; ++j) { float v = sQ[tid * 256 + j]; s += v * v; }
    sNrm[tid] = fmaxf(sqrtf(s), 1e-12f);
  }
  __syncthreads();
  for (int i = tid; i < 16 * 256; i += 256) {
    int m = i >> 8;
    q_in[row0 * 256 + i] = sQ[i] / sNrm[m];
  }
}

// ---------------------------------------------------------------- K6: 8-qubit, 7-layer circuit; one wave32 per sample, state in registers
__global__ __launch_bounds__(256) void hq_quantum(const float* __restrict__ q_in,
                                                  const float* __restrict__ qw,
                                                  float* __restrict__ q_raw) {
  int wave = threadIdx.x >> 5, lane = threadIdx.x & 31;
  int s = blockIdx.x * 8 + wave;
  // amplitude index = lane*8 + r : bits 7..3 = lane, 2..0 = r (bit 7 = qubit 0)
  float amp[8];
  const float4* p4 = (const float4*)(q_in + s * 256 + lane * 8);
  float4 v0 = p4[0], v1 = p4[1];
  amp[0] = v0.x; amp[1] = v0.y; amp[2] = v0.z; amp[3] = v0.w;
  amp[4] = v1.x; amp[5] = v1.y; amp[6] = v1.z; amp[7] = v1.w;

  for (int layer = 0; layer < 7; ++layer) {
    const float* w = qw + layer * 8;
    // RY qubits 0..4 -> lane bits 4..0 (cross-lane)
#pragma unroll
    for (int i = 0; i < 5; ++i) {
      float th = w[i] * 0.5f;
      float cs = cosf(th), sn = sinf(th);
      int lb = 4 - i, mask = 1 << lb;
      float sgn = ((lane >> lb) & 1) ? sn : -sn;
#pragma unroll
      for (int r = 0; r < 8; ++r) {
        float p = __shfl_xor(amp[r], mask, 32);
        amp[r] = fmaf(sgn, p, cs * amp[r]);
      }
    }
    // RY qubits 5..7 -> register bits 2..0 (in-lane)
#pragma unroll
    for (int i = 5; i < 8; ++i) {
      float th = w[i] * 0.5f;
      float cs = cosf(th), sn = sinf(th);
      int bb = 7 - i;
#pragma unroll
      for (int r = 0; r < 8; ++r) {
        if (!((r >> bb) & 1)) {
          int r1 = r | (1 << bb);
          float a0 = amp[r], a1 = amp[r1];
          amp[r]  = cs * a0 - sn * a1;
          amp[r1] = sn * a0 + cs * a1;
        }
      }
    }
    // CNOT (q0,q1)..(q3,q4): ctrl lane bit 4..1, tgt lane bit 3..0
#pragma unroll
    for (int j = 0; j < 4; ++j) {
      int cb = 4 - j, tmask = 1 << (3 - j);
      bool ctrl = (lane >> cb) & 1;
#pragma unroll
      for (int r = 0; r < 8; ++r) {
        float p = __shfl_xor(amp[r], tmask, 32);
        amp[r] = ctrl ? p : amp[r];
      }
    }
    // CNOT (q4,q5): ctrl lane bit0, tgt reg bit2
    if (lane & 1) {
#pragma unroll
      for (int r = 0; r < 4; ++r) { float t = amp[r]; amp[r] = amp[r + 4]; amp[r + 4] = t; }
    }
    // CNOT (q5,q6): ctrl reg bit2, tgt reg bit1 -> swap (4,6),(5,7)
    { float t = amp[4]; amp[4] = amp[6]; amp[6] = t; t = amp[5]; amp[5] = amp[7]; amp[7] = t; }
    // CNOT (q6,q7): ctrl reg bit1, tgt reg bit0 -> swap (2,3),(6,7)
    { float t = amp[2]; amp[2] = amp[3]; amp[3] = t; t = amp[6]; amp[6] = amp[7]; amp[7] = t; }
    // CNOT (q7,q0): ctrl reg bit0, tgt lane bit4
#pragma unroll
    for (int r = 1; r < 8; r += 2) amp[r] = __shfl_xor(amp[r], 16, 32);
  }
  // measure Z on qubits 0..3 (lane bits 4..1)
  float p = 0.f;
#pragma unroll
  for (int r = 0; r < 8; ++r) p = fmaf(amp[r], amp[r], p);
  float z[4];
#pragma unroll
  for (int q = 0; q < 4; ++q) z[q] = ((lane >> (4 - q)) & 1) ? -p : p;
#pragma unroll
  for (int off = 16; off > 0; off >>= 1)
#pragma unroll
    for (int q = 0; q < 4; ++q) z[q] += __shfl_xor(z[q], off, 32);
  if (lane == 0) {
#pragma unroll
    for (int q = 0; q < 4; ++q) q_raw[s * 4 + q] = z[q];
  }
}

// ---------------------------------------------------------------- K7a: post1 + relu + bnp stats
__global__ __launch_bounds__(256) void hq_post1(const float* __restrict__ q_raw,
                                                const float* __restrict__ w,
                                                const float* __restrict__ b,
                                                float* __restrict__ h2pre, float* stats) {
  __shared__ float sS[32], sQ[32];
  int tid = threadIdx.x;
  if (tid < 32) { sS[tid] = 0.f; sQ[tid] = 0.f; }
  __syncthreads();
  int row = blockIdx.x * 256 + tid;
  float q0 = q_raw[row * 4 + 0], q1 = q_raw[row * 4 + 1];
  float q2 = q_raw[row * 4 + 2], q3 = q_raw[row * 4 + 3];
#pragma unroll
  for (int j = 0; j < 32; ++j) {
    float v = b[j] + q0 * w[j * 4] + q1 * w[j * 4 + 1] + q2 * w[j * 4 + 2] + q3 * w[j * 4 + 3];
    v = fmaxf(v, 0.f);
    h2pre[row * 32 + j] = v;
    atomicAdd(&sS[j], v);
    atomicAdd(&sQ[j], v * v);
  }
  __syncthreads();
  if (tid < 32) { atomicAdd(&stats[SPS + tid], sS[tid]); atomicAdd(&stats[SPQ + tid], sQ[tid]); }
}

// ---------------------------------------------------------------- K7b: bnp -> post2 tanh -> concat -> fus1 + bnf stats
__global__ __launch_bounds__(256) void hq_fuse1(const float* __restrict__ h2pre,
                                                const float* __restrict__ stats,
                                                const float* __restrict__ gp,
                                                const float* __restrict__ bp,
                                                const float* __restrict__ w2,
                                                const float* __restrict__ b2,
                                                const float* __restrict__ cnn_feat,
                                                const float* __restrict__ w1,
                                                const float* __restrict__ b1,
                                                float* __restrict__ f1, float* statsO) {
  __shared__ float sc[32], sh[32], sS[32], sQ[32];
  int tid = threadIdx.x;
  if (tid < 32) {
    float m = stats[SPS + tid] / 4096.f;
    float v = stats[SPQ + tid] / 4096.f - m * m;
    float s = gp[tid] * rsqrtf(v + EPS_BN);
    sc[tid] = s; sh[tid] = bp[tid] - m * s;
    sS[tid] = 0.f; sQ[tid] = 0.f;
  }
  __syncthreads();
  int row = blockIdx.x * 256 + tid;
  float h2[32], fused[48];
#pragma unroll
  for (int k = 0; k < 32; ++k) h2[k] = fmaf(h2pre[row * 32 + k], sc[k], sh[k]);
#pragma unroll
  for (int k = 0; k < 32; ++k) fused[k] = cnn_feat[row * 32 + k];
#pragma unroll
  for (int j = 0; j < 16; ++j) {
    float a = b2[j];
#pragma unroll
    for (int k = 0; k < 32; ++k) a = fmaf(h2[k], w2[j * 32 + k], a);
    fused[32 + j] = tanhf(a);
  }
#pragma unroll
  for (int j = 0; j < 32; ++j) {
    float a = b1[j];
#pragma unroll
    for (int k = 0; k < 48; ++k) a = fmaf(fused[k], w1[j * 48 + k], a);
    f1[row * 32 + j] = a;
    atomicAdd(&sS[j], a);
    atomicAdd(&sQ[j], a * a);
  }
  __syncthreads();
  if (tid < 32) { atomicAdd(&statsO[SFS + tid], sS[tid]); atomicAdd(&statsO[SFQ + tid], sQ[tid]); }
}

// ---------------------------------------------------------------- K7c: bnf -> relu -> fus2 relu -> fus3 sigmoid
__global__ __launch_bounds__(256) void hq_fuse2(const float* __restrict__ f1,
                                               const float* __restrict__ stats,
                                               const float* __restrict__ gf,
                                               const float* __restrict__ bf,
                                               const float* __restrict__ w2,
                                               const float* __restrict__ b2,
                                               const float* __restrict__ w3,
                                               const float* __restrict__ b3,
                                               float* __restrict__ out) {
  __shared__ float sc[32], sh[32];
  int tid = threadIdx.x;
  if (tid < 32) {
    float m = stats[SFS + tid] / 4096.f;
    float v = stats[SFQ + tid] / 4096.f - m * m;
    float s = gf[tid] * rsqrtf(v + EPS_BN);
    sc[tid] = s; sh[tid] = bf[tid] - m * s;
  }
  __syncthreads();
  int row = blockIdx.x * 256 + tid;
  float h3[32], h4[16];
#pragma unroll
  for (int k = 0; k < 32; ++k) h3[k] = fmaxf(fmaf(f1[row * 32 + k], sc[k], sh[k]), 0.f);
#pragma unroll
  for (int j = 0; j < 16; ++j) {
    float a = b2[j];
#pragma unroll
    for (int k = 0; k < 32; ++k) a = fmaf(h3[k], w2[j * 32 + k], a);
    h4[j] = fmaxf(a, 0.f);
  }
  float z = b3[0];
#pragma unroll
  for (int k = 0; k < 16; ++k) z = fmaf(h4[k], w3[k], z);
  out[row] = 1.f / (1.f + expf(-z));
}

// ---------------------------------------------------------------- launch
extern "C" void kernel_launch(void* const* d_in, const int* in_sizes, int n_in,
                              void* d_out, int out_size, void* d_ws, size_t ws_size,
                              hipStream_t stream) {
  const float* x     = (const float*)d_in[0];
  const float* c1w   = (const float*)d_in[1];
  const float* c1b   = (const float*)d_in[2];
  const float* g1    = (const float*)d_in[3];
  const float* bb1   = (const float*)d_in[4];
  const float* c2w   = (const float*)d_in[5];
  const float* c2b   = (const float*)d_in[6];
  const float* g2    = (const float*)d_in[7];
  const float* bb2   = (const float*)d_in[8];
  const float* c3w   = (const float*)d_in[9];
  const float* c3b   = (const float*)d_in[10];
  const float* g3    = (const float*)d_in[11];
  const float* bb3   = (const float*)d_in[12];
  const float* fcw   = (const float*)d_in[13];
  const float* fcb   = (const float*)d_in[14];
  const float* p1w   = (const float*)d_in[15];
  const float* p1b   = (const float*)d_in[16];
  const float* p2w   = (const float*)d_in[17];
  const float* p2b   = (const float*)d_in[18];
  const float* qwts  = (const float*)d_in[19];
  const float* po1w  = (const float*)d_in[20];
  const float* po1b  = (const float*)d_in[21];
  const float* gp    = (const float*)d_in[22];
  const float* bp    = (const float*)d_in[23];
  const float* po2w  = (const float*)d_in[24];
  const float* po2b  = (const float*)d_in[25];
  const float* f1w   = (const float*)d_in[26];
  const float* f1b   = (const float*)d_in[27];
  const float* gf    = (const float*)d_in[28];
  const float* bf    = (const float*)d_in[29];
  const float* f2w   = (const float*)d_in[30];
  const float* f2b   = (const float*)d_in[31];
  const float* f3w   = (const float*)d_in[32];
  const float* f3b   = (const float*)d_in[33];

  float* ws    = (float*)d_ws;
  float* stats = ws;
  float* c1    = ws + OFF_C1;
  float* c2    = ws + OFF_C2;
  float* c3    = ws + OFF_C3;
  float* cf    = ws + OFF_CF;
  float* qi    = ws + OFF_QI;
  float* qr    = ws + OFF_QR;
  float* h2    = ws + OFF_H2;
  float* f1    = ws + OFF_F1;
  float* out   = (float*)d_out;

  hq_zero_stats<<<1, 256, 0, stream>>>(stats);
  hq_conv1<<<12544, 256, 0, stream>>>(x, c1w, c1b, c1, stats);
  hq_conv2<<<6272, 256, 0, stream>>>(c1, c2w, c2b, g1, bb1, c2, stats);
  hq_conv3<<<2304, 256, 0, stream>>>(c2, c3w, c3b, g2, bb2, c3, stats);
  hq_cnn_head<<<64, 128, 0, stream>>>(c3, stats, g3, bb3, fcw, fcb, cf);
  hq_pre_path<<<256, 256, 0, stream>>>(x, p1w, p1b, p2w, p2b, qi);
  hq_quantum<<<512, 256, 0, stream>>>(qi, qwts, qr);
  hq_post1<<<16, 256, 0, stream>>>(qr, po1w, po1b, h2, stats);
  hq_fuse1<<<16, 256, 0, stream>>>(h2, stats, gp, bp, po2w, po2b, cf, f1w, f1b, f1, stats);
  hq_fuse2<<<16, 256, 0, stream>>>(f1, stats, gf, bf, f2w, f2b, f3w, f3b, out);
}
